// FrameFusion_412316861057
// MI455X (gfx1250) — matrix operand order repaired and checked by reading, not compiled, not run on hardware
//
#include <hip/hip_runtime.h>
#include <hip/hip_bf16.h>
#include <cstdint>

typedef __attribute__((ext_vector_type(16))) _Float16 v16h;
typedef __attribute__((ext_vector_type(8)))  float    v8f;
typedef int v4i __attribute__((vector_size(16)));

constexpr int B = 2, N = 4096, D = 1024, T = 1024;

#if defined(__has_builtin)
#if __has_builtin(__builtin_amdgcn_global_load_async_to_lds_b128)
#define HAVE_ASYNC_LDS 1
#endif
#endif

#define GAS_V4I __attribute__((address_space(1))) v4i*
#define LAS_V4I __attribute__((address_space(3))) v4i*

__device__ __forceinline__ void wait_async_lds() {
#if defined(HAVE_ASYNC_LDS)
#if __has_builtin(__builtin_amdgcn_s_wait_asynccnt)
    __builtin_amdgcn_s_wait_asynccnt(0);
#else
    asm volatile("s_wait_asynccnt 0" ::: "memory");
#endif
#endif
}

// ---------------- global min/max of attention -> r scale params ----------------
__global__ void dpp_stat_kernel(const float* __restrict__ att, float* __restrict__ stat) {
    __shared__ float smin[256], smax[256];
    int t = threadIdx.x;
    float mn = 1e30f, mx = -1e30f;
    for (int i = t; i < B * N; i += 256) { float v = att[i]; mn = fminf(mn, v); mx = fmaxf(mx, v); }
    smin[t] = mn; smax[t] = mx; __syncthreads();
    for (int s = 128; s > 0; s >>= 1) {
        if (t < s) { smin[t] = fminf(smin[t], smin[t + s]); smax[t] = fmaxf(smax[t], smax[t + s]); }
        __syncthreads();
    }
    if (t == 0) { stat[0] = smax[0]; stat[1] = 1.0f / (smax[0] - smin[0]); }
}

// r = (att_max - att + 1e-6) / (att_max - att_min);  di2s0 = r^2 (sim_ii == 1)
__global__ void dpp_rinit_kernel(const float* __restrict__ att, const float* __restrict__ stat,
                                 float* __restrict__ r, float* __restrict__ di2s) {
    int i = blockIdx.x * 256 + threadIdx.x;
    float rv = (stat[0] - att[i] + 1e-6f) * stat[1];
    r[i] = rv; di2s[i] = rv * rv;
}

// ---------------- row-normalize features, cast to f16 ----------------
__global__ void dpp_norm_kernel(const float* __restrict__ x, _Float16* __restrict__ xh) {
    __shared__ float red[256];
    int row = blockIdx.x, t = threadIdx.x;
    const float* xr = x + (size_t)row * D;
    float s = 0.f;
    for (int k = t; k < D; k += 256) { float v = xr[k]; s += v * v; }
    red[t] = s; __syncthreads();
    for (int st = 128; st > 0; st >>= 1) { if (t < st) red[t] += red[t + st]; __syncthreads(); }
    float inv = rsqrtf(red[0]);
    _Float16* xo = xh + (size_t)row * D;
    for (int k = t; k < D; k += 256) xo[k] = (_Float16)(xr[k] * inv);
}

// ---------------- WMMA GEMM: K[b,m,n] = r_m * r_n * <x_m, x_n> ----------------
// block tile 128(M) x 128(N), 8 waves (4Mx2N), wave tile 32x64 = 2x4 WMMA tiles.
// Double-buffered LDS; tiles fetched with GLOBAL_LOAD_ASYNC_TO_LDS_B128 when available.
constexpr int AB_HALVES = 128 * 32;          // one A (or B) tile: 8 KB
constexpr int BUF_HALVES = 2 * AB_HALVES;    // A + B per stage: 16 KB

__global__ void __launch_bounds__(256)
dpp_gemm_kernel(const _Float16* __restrict__ xh, const float* __restrict__ r,
                float* __restrict__ Kmat) {
    __shared__ _Float16 lds[2 * BUF_HALVES]; // 32 KB (two stages)
    const int b = blockIdx.z;
    const int nBase = blockIdx.x * 128;
    const int mBase = blockIdx.y * 128;
    const int tid = threadIdx.x;
    const int lane = tid & 31;
    const int w = tid >> 5;
    const int wm = (w >> 1) * 32;   // 0,32,64,96
    const int wn = (w & 1) * 64;    // 0,64
    const _Float16* xb = xh + (size_t)b * N * D;

    // per-thread tile-fetch addresses: 2 threads/row, 32B each, for A and B tiles
    const int rrow = tid >> 1, part = tid & 1;
    const _Float16* gA = xb + (size_t)(mBase + rrow) * D + part * 16;
    const _Float16* gB = xb + (size_t)(nBase + rrow) * D + part * 16;
    _Float16* lA = lds + rrow * 32 + part * 16;
    _Float16* lB = lA + AB_HALVES;

    auto issueTile = [&](int k0, int buf) {
        const _Float16* sa = gA + k0;
        const _Float16* sb = gB + k0;
        _Float16* da = lA + buf * BUF_HALVES;
        _Float16* db = lB + buf * BUF_HALVES;
#if defined(HAVE_ASYNC_LDS)
        __builtin_amdgcn_global_load_async_to_lds_b128((GAS_V4I)sa, (LAS_V4I)da, 0, 0);
        __builtin_amdgcn_global_load_async_to_lds_b128((GAS_V4I)sa, (LAS_V4I)da, 16, 0);
        __builtin_amdgcn_global_load_async_to_lds_b128((GAS_V4I)sb, (LAS_V4I)db, 0, 0);
        __builtin_amdgcn_global_load_async_to_lds_b128((GAS_V4I)sb, (LAS_V4I)db, 16, 0);
#else
        ((uint4*)da)[0] = ((const uint4*)sa)[0];
        ((uint4*)da)[1] = ((const uint4*)sa)[1];
        ((uint4*)db)[0] = ((const uint4*)sb)[0];
        ((uint4*)db)[1] = ((const uint4*)sb)[1];
#endif
    };

    v8f acc[2][4] = {};
    const int hi = lane >> 4, lo = lane & 15;

    issueTile(0, 0);
    wait_async_lds();
    __syncthreads();

    int buf = 0;
    for (int k0 = 0; k0 < D; k0 += 32) {
        if (k0 + 32 < D) issueTile(k0 + 32, buf ^ 1);   // overlap DMA with WMMA

        const _Float16* As = lds + buf * BUF_HALVES;
        const _Float16* Bs = As + AB_HALVES;
        v16h afrag[2], bfrag[4];
        #pragma unroll
        for (int tm = 0; tm < 2; ++tm) {   // A 16x32: lanes0-15 K=0-15; lanes16-31 K=16-31
            const _Float16* ap = As + (wm + tm * 16 + lo) * 32;
            uint4* q = (uint4*)&afrag[tm];
            q[0] = *(const uint4*)(ap + hi * 16);
            q[1] = *(const uint4*)(ap + hi * 16 + 8);
        }
        #pragma unroll
        for (int tn = 0; tn < 4; ++tn) {   // B 32x16: lanes0-15 K=0-15; lanes16-31 K=16-31
            const _Float16* bp = Bs + (wn + tn * 16 + lo) * 32 + hi * 16;
            uint4* q = (uint4*)&bfrag[tn];
            q[0] = ((const uint4*)bp)[0];
            q[1] = ((const uint4*)bp)[1];
        }
        #pragma unroll
        for (int tm = 0; tm < 2; ++tm)
            #pragma unroll
            for (int tn = 0; tn < 4; ++tn)
                acc[tm][tn] = __builtin_amdgcn_wmma_f32_16x16x32_f16(
                    false, afrag[tm], false, bfrag[tn], (short)0, acc[tm][tn], false, false);

        wait_async_lds();
        __syncthreads();
        buf ^= 1;
    }

    // epilogue: D layout lane->N=lane&15, VGPR v -> M = v + 8*(lane>=16)
    const float* rb = r + b * N;
    #pragma unroll
    for (int tm = 0; tm < 2; ++tm)
        #pragma unroll
        for (int tn = 0; tn < 4; ++tn) {
            int gn = nBase + wn + tn * 16 + lo;
            float rn = rb[gn];
            #pragma unroll
            for (int v = 0; v < 8; ++v) {
                int gm = mBase + wm + tm * 16 + hi * 8 + v;
                Kmat[((size_t)b * N + gm) * N + gn] = acc[tm][tn][v] * rn * rb[gm];
            }
        }
}

// ---------------- per-step argmax over di2s ----------------
__global__ void dpp_select_kernel(const float* __restrict__ di2s, int* __restrict__ jbuf,
                                  float* __restrict__ isd, int* __restrict__ outSel, int step) {
    __shared__ float sv[256]; __shared__ int si[256];
    int b = blockIdx.x, t = threadIdx.x;
    const float* d = di2s + b * N;
    float best = -__builtin_inff(); int bi = 0;
    for (int n = t; n < N; n += 256) { float v = d[n]; if (v > best) { best = v; bi = n; } }
    sv[t] = best; si[t] = bi; __syncthreads();
    for (int s = 128; s > 0; s >>= 1) {
        if (t < s) {
            if (sv[t + s] > sv[t] || (sv[t + s] == sv[t] && si[t + s] < si[t])) {
                sv[t] = sv[t + s]; si[t] = si[t + s];
            }
        }
        __syncthreads();
    }
    if (t == 0) {
        jbuf[b] = si[0];
        isd[b] = rsqrtf(sv[0]);
        if (b == 0) outSel[step] = si[0];
    }
}

// ---------------- per-step Cholesky-row update ----------------
__global__ void __launch_bounds__(256)
dpp_update_kernel(const float* __restrict__ Kmat, float* __restrict__ cis,
                  float* __restrict__ di2s, const int* __restrict__ jbuf,
                  const float* __restrict__ isd, int step) {
    __shared__ float cj[256];
    int b = blockIdx.y;
    int n = blockIdx.x * 256 + threadIdx.x;
    int j = jbuf[b];
    float s = isd[b];
    float kj = Kmat[((size_t)b * N + j) * N + n];
    float acc = 0.f;
    for (int t0 = 0; t0 < step; t0 += 256) {
        int tt = t0 + threadIdx.x;
        cj[threadIdx.x] = (tt < step) ? cis[((size_t)tt * B + b) * N + j] : 0.f;
        __syncthreads();
        if (t0 + 256 < step)
            __builtin_prefetch(&cis[((size_t)(t0 + 256) * B + b) * N + n], 0, 0);
        int lim = (step - t0 < 256) ? (step - t0) : 256;
        for (int u = 0; u < lim; ++u)
            acc = fmaf(cj[u], cis[((size_t)(t0 + u) * B + b) * N + n], acc);
        __syncthreads();
    }
    float eis = (kj - acc) * s;
    cis[((size_t)step * B + b) * N + n] = eis;
    float d2 = di2s[b * N + n] - eis * eis;
    if (n == j) d2 = -__builtin_inff();
    di2s[b * N + n] = d2;
}

extern "C" void kernel_launch(void* const* d_in, const int* in_sizes, int n_in,
                              void* d_out, int out_size, void* d_ws, size_t ws_size,
                              hipStream_t stream) {
    const float* feat = (const float*)d_in[0];   // [B,N,D] fp32
    const float* att  = (const float*)d_in[1];   // [B,N]   fp32
    (void)in_sizes; (void)n_in; (void)out_size; (void)ws_size;
    int* outSel = (int*)d_out;                   // [T] int32 (batch-0 selections)

    char* ws = (char*)d_ws;
    float*    Kmat = (float*)ws;     ws += (size_t)B * N * N * sizeof(float);    // 134 MB
    float*    cis  = (float*)ws;     ws += (size_t)T * B * N * sizeof(float);    // 33.5 MB
    _Float16* xh   = (_Float16*)ws;  ws += (size_t)B * N * D * sizeof(_Float16); // 16.8 MB
    float*    r    = (float*)ws;     ws += (size_t)B * N * sizeof(float);
    float*    di2s = (float*)ws;     ws += (size_t)B * N * sizeof(float);
    float*    stat = (float*)ws;     ws += 256;
    int*      jbuf = (int*)ws;       ws += 256;
    float*    isd  = (float*)ws;     ws += 256;

    dpp_stat_kernel <<<1, 256, 0, stream>>>(att, stat);
    dpp_rinit_kernel<<<(B * N) / 256, 256, 0, stream>>>(att, stat, r, di2s);
    dpp_norm_kernel <<<B * N, 256, 0, stream>>>(feat, xh);
    dpp_gemm_kernel <<<dim3(N / 128, N / 128, B), 256, 0, stream>>>(xh, r, Kmat);

    for (int i = 0; i < T; ++i) {
        dpp_select_kernel<<<B, 256, 0, stream>>>(di2s, jbuf, isd, outSel, i);
        dpp_update_kernel<<<dim3(N / 256, B), 256, 0, stream>>>(Kmat, cis, di2s, jbuf, isd, i);
    }
}
